// RNNDecoder_52853867545205
// MI455X (gfx1250) — matrix operand order repaired and checked by reading, not compiled, not run on hardware
//
#include <hip/hip_runtime.h>
#include <hip/hip_bf16.h>
#include <math.h>

// Problem dims (fixed by reference setup_inputs)
#define B_ 32
#define T_ 64
#define S_ 256
#define H_ 1024
#define E_ 512

typedef __attribute__((ext_vector_type(16))) __bf16 v16bf;
typedef __attribute__((ext_vector_type(8)))  __bf16 v8bf;
typedef __attribute__((ext_vector_type(8)))  float  v8f;

__device__ __forceinline__ float sigmoidf_(float x) {
    return 1.0f / (1.0f + __expf(-x));
}

#define SHUF16(lo, hi) __builtin_shufflevector(lo, hi, \
        0, 1, 2, 3, 4, 5, 6, 7, 8, 9, 10, 11, 12, 13, 14, 15)

// ---------------------------------------------------------------------------
// WMMA GEMM:  C[M=32, N] = A[32, K] (bf16) x W[N, K] (bf16, row-major) ^T
// Block = 256 threads = 8 waves.  Each wave computes TWO adjacent 16x16 C
// tiles (A-fragment reuse), so a block covers a 32x128 tile of C.
// K must be a multiple of 64.  The K-loop is a copy-free two-stage ping-pong:
// stage0 fragments cover K=k0, stage1 K=k0+32; each WMMA pair consumes one
// named variable set while loads fill the other, so no register rotation is
// required and pre-WMMA waits only cover loads issued one WMMA pair (2 WMMAs) earlier.
// Fragment layouts used here (wave32; A/B K order is contiguous, not the ISA table order):
//   A 16x32 bf16 : lane%16 = M; lanes 0-15 hold K 0..15,
//                  lanes 16-31 hold K 16..31, 2 per VGPR.
//   B 32x16 bf16 : lane%16 = N; lanes 0-15 hold K 0..15,
//                  lanes 16-31 hold K 16..31 (contiguous along K in memory).
//   C 16x16 f32  : VGPR i -> M = (lane/16)*8 + i, N = lane%16.
// ---------------------------------------------------------------------------
__global__ void __launch_bounds__(256)
gemm_bf16_wmma(const __bf16* __restrict__ A, const __bf16* __restrict__ W,
               float* __restrict__ C, int N, int K)
{
    const int wave  = threadIdx.x >> 5;
    const int lane  = threadIdx.x & 31;
    const int half  = lane >> 4;          // 16-lane group
    const int l16   = lane & 15;
    const int mbase = (wave >> 2) << 4;                    // 0 or 16
    const int nbase = blockIdx.x * 128 + (wave & 3) * 32;  // two 16-wide tiles

    v8f acc0 = {}, acc1 = {};
    const __bf16* arow  = A + (size_t)(mbase + l16) * K + half * 16;
    const __bf16* wrow0 = W + (size_t)(nbase + l16) * K + half * 16;
    const __bf16* wrow1 = wrow0 + (size_t)16 * K;

    // Pipeline prologue: stage0 fragments for k0 = 0.
    v8bf  alo0 = *(const v8bf*)(arow);
    v8bf  ahi0 = *(const v8bf*)(arow + 8);
    v16bf b00  = *(const v16bf*)(wrow0);
    v16bf b10  = *(const v16bf*)(wrow1);

    for (int k0 = 0; k0 < K; k0 += 64) {
        // Stage1 loads (k0+32 always < K since K % 64 == 0).
        v8bf  alo1 = *(const v8bf*)(arow  + k0 + 32);
        v8bf  ahi1 = *(const v8bf*)(arow  + k0 + 40);
        v16bf b01  = *(const v16bf*)(wrow0 + k0 + 32);
        v16bf b11  = *(const v16bf*)(wrow1 + k0 + 32);
        if (k0 + 512 < K) {  // WGP-scope stream-ahead on the weight rows
            asm volatile("global_prefetch_b8 %0, off" :: "v"(wrow0 + k0 + 512));
            asm volatile("global_prefetch_b8 %0, off" :: "v"(wrow1 + k0 + 512));
        }

        v16bf a0 = SHUF16(alo0, ahi0);
        acc0 = __builtin_amdgcn_wmma_f32_16x16x32_bf16(
                   false, a0, false, b00, (short)0, acc0, false, false);
        acc1 = __builtin_amdgcn_wmma_f32_16x16x32_bf16(
                   false, a0, false, b10, (short)0, acc1, false, false);

        // Refill stage0 for k0+64 (skipped only on the final iteration).
        if (k0 + 64 < K) {
            alo0 = *(const v8bf*)(arow  + k0 + 64);
            ahi0 = *(const v8bf*)(arow  + k0 + 72);
            b00  = *(const v16bf*)(wrow0 + k0 + 64);
            b10  = *(const v16bf*)(wrow1 + k0 + 64);
        }

        v16bf a1 = SHUF16(alo1, ahi1);
        acc0 = __builtin_amdgcn_wmma_f32_16x16x32_bf16(
                   false, a1, false, b01, (short)0, acc0, false, false);
        acc1 = __builtin_amdgcn_wmma_f32_16x16x32_bf16(
                   false, a1, false, b11, (short)0, acc1, false, false);
    }
    #pragma unroll
    for (int i = 0; i < 8; ++i) {
        C[(size_t)(mbase + half * 8 + i) * N + nbase + l16]      = acc0[i];
        C[(size_t)(mbase + half * 8 + i) * N + nbase + 16 + l16] = acc1[i];
    }
}

// ---------------------------------------------------------------------------
// One-time prep kernels
// ---------------------------------------------------------------------------

// Concat two fp32 weight matrices [N,k0],[N,k1] along K and convert to bf16
// row-major [N, k0+k1].  k1 may be 0 (w1 unused).
__global__ void __launch_bounds__(256)
cvt_cat_bf16(const float* __restrict__ w0, int k0,
             const float* __restrict__ w1, int k1,
             __bf16* __restrict__ dst, long total)
{
    const int K = k0 + k1;
    for (long idx = blockIdx.x * 256L + threadIdx.x; idx < total;
         idx += (long)gridDim.x * 256L) {
        long n = idx / K;
        int  k = (int)(idx % K);
        float v = (k < k0) ? w0[n * (long)k0 + k]
                           : w1[n * (long)k1 + (k - k0)];
        dst[idx] = (__bf16)v;
    }
}

// emb_all[b,t,e] = bf16(embedding[tgt[b,t], e])
__global__ void __launch_bounds__(256)
gather_emb(const int* __restrict__ tgt, const float* __restrict__ table,
           __bf16* __restrict__ emb_all)
{
    int idx = blockIdx.x * 256 + threadIdx.x;   // B_*T_*E_ exact
    int bt = idx >> 9;                          // E_ = 512
    int e  = idx & 511;
    emb_all[idx] = (__bf16)table[(long)tgt[bt] * E_ + e];
}

__global__ void compute_len(const int* __restrict__ masks, int* __restrict__ len)
{
    int b = threadIdx.x;
    if (b < B_) {
        int s = 0;
        for (int j = 0; j < S_; ++j) s += masks[b * S_ + j];
        len[b] = s;
    }
}

// h0s = bf16(h_in[0]);  xcat1[:,1024:2048] = bf16(h_in[1])
__global__ void __launch_bounds__(256)
init_states(const float* __restrict__ h_in, __bf16* __restrict__ h0s,
            __bf16* __restrict__ xcat1)
{
    int idx = blockIdx.x * 256 + threadIdx.x;   // B_*H_ exact
    int b = idx >> 10, h = idx & 1023;
    h0s[idx] = (__bf16)h_in[idx];
    xcat1[b * 2048 + 1024 + h] = (__bf16)h_in[B_ * H_ + idx];
}

// ---------------------------------------------------------------------------
// Per-step kernels
// ---------------------------------------------------------------------------

// xcat0 = [emb_t (512) | feed (1024) | h0 (1024)]  -> bf16 [32, 2560]
__global__ void __launch_bounds__(256)
pack_xcat0(const __bf16* __restrict__ emb_all, const __bf16* __restrict__ feed,
           const __bf16* __restrict__ h0, __bf16* __restrict__ xcat0, int t)
{
    int idx = blockIdx.x * 256 + threadIdx.x;   // B_*2560 exact
    int b = idx / 2560, k = idx % 2560;
    __bf16 v;
    if (k < 512)        v = emb_all[((long)b * T_ + t) * E_ + k];
    else if (k < 1536)  v = feed[b * H_ + (k - 512)];
    else                v = h0[b * H_ + (k - 1536)];
    xcat0[idx] = v;
}

// LSTM cell, layer 0: gates [B,4096] (fp32) in (i,f,g,o) blocks of H.
// c stays fp32; h emitted as bf16 to h0 state buffer and xcat1[:,0:1024].
__global__ void __launch_bounds__(256)
lstm_cell0(const float* __restrict__ gates, const float* __restrict__ b_ih,
           const float* __restrict__ b_hh, float* __restrict__ c,
           __bf16* __restrict__ hbuf, __bf16* __restrict__ xcat1)
{
    int idx = blockIdx.x * 256 + threadIdx.x;   // B_*H_ exact
    int b = idx >> 10, h = idx & 1023;
    const float* g = gates + b * 4096;
    float gi = g[h]        + b_ih[h]        + b_hh[h];
    float gf = g[h + 1024] + b_ih[h + 1024] + b_hh[h + 1024];
    float gg = g[h + 2048] + b_ih[h + 2048] + b_hh[h + 2048];
    float go = g[h + 3072] + b_ih[h + 3072] + b_hh[h + 3072];
    float cn = sigmoidf_(gf) * c[idx] + sigmoidf_(gi) * tanhf(gg);
    float hn = sigmoidf_(go) * tanhf(cn);
    c[idx] = cn;
    __bf16 hb = (__bf16)hn;
    hbuf[idx] = hb;
    xcat1[b * 2048 + h] = hb;
}

// LSTM cell, layer 1: h1 -> h1b (for q GEMM), xcat1[:,1024:] (next step),
// ctxcat[:,1024:] (this step's output concat).  All bf16.
__global__ void __launch_bounds__(256)
lstm_cell1(const float* __restrict__ gates, const float* __restrict__ b_ih,
           const float* __restrict__ b_hh, float* __restrict__ c,
           __bf16* __restrict__ h1b, __bf16* __restrict__ xcat1,
           __bf16* __restrict__ ctxcat)
{
    int idx = blockIdx.x * 256 + threadIdx.x;
    int b = idx >> 10, h = idx & 1023;
    const float* g = gates + b * 4096;
    float gi = g[h]        + b_ih[h]        + b_hh[h];
    float gf = g[h + 1024] + b_ih[h + 1024] + b_hh[h + 1024];
    float gg = g[h + 2048] + b_ih[h + 2048] + b_hh[h + 2048];
    float go = g[h + 3072] + b_ih[h + 3072] + b_hh[h + 3072];
    float cn = sigmoidf_(gf) * c[idx] + sigmoidf_(gi) * tanhf(gg);
    float hn = sigmoidf_(go) * tanhf(cn);
    c[idx] = cn;
    __bf16 hb = (__bf16)hn;
    h1b[idx] = hb;
    xcat1[b * 2048 + 1024 + h]  = hb;
    ctxcat[b * 2048 + 1024 + h] = hb;
}

// scores[b,s] = dot(q[b,:], mem[b,s,:]) with length mask.
// One wave per (b,s); cross-lane reduce via shfl_xor (wave32).
__global__ void __launch_bounds__(256)
attn_scores(const float* __restrict__ q, const float* __restrict__ mem,
            const int* __restrict__ len, float* __restrict__ scores)
{
    const int wave = threadIdx.x >> 5;
    const int lane = threadIdx.x & 31;
    const int gi = blockIdx.x * 8 + wave;   // B_*S_ waves total
    const int b = gi >> 8;
    const int s = gi & 255;
    const float* m  = mem + ((size_t)b * S_ + s) * H_;
    const float* qq = q + b * H_;
    float acc = 0.f;
    #pragma unroll 4
    for (int j = lane; j < H_; j += 32) acc += m[j] * qq[j];
    #pragma unroll
    for (int off = 16; off > 0; off >>= 1) acc += __shfl_xor(acc, off, 32);
    if (lane == 0) scores[b * S_ + s] = (s < len[b]) ? acc : -1e9f;
}

// Row softmax over S=256; writes attn workspace and d_out attention slice.
__global__ void __launch_bounds__(256)
attn_softmax(const float* __restrict__ scores, float* __restrict__ attn,
             float* __restrict__ out_attn, int t)
{
    __shared__ float red[256];
    const int b = blockIdx.x, s = threadIdx.x;
    float v = scores[b * S_ + s];
    red[s] = v; __syncthreads();
    for (int o = 128; o > 0; o >>= 1) {
        if (s < o) red[s] = fmaxf(red[s], red[s + o]);
        __syncthreads();
    }
    float mx = red[0]; __syncthreads();
    float e = __expf(v - mx);
    red[s] = e; __syncthreads();
    for (int o = 128; o > 0; o >>= 1) {
        if (s < o) red[s] += red[s + o];
        __syncthreads();
    }
    float a = e / red[0];
    attn[b * S_ + s] = a;
    out_attn[((long)b * T_ + t) * S_ + s] = a;
}

// ctxcat[:,0:1024] = bf16(attn @ memory_bank)  (per-batch weighted sum over S)
__global__ void __launch_bounds__(256)
attn_ctx(const float* __restrict__ attn, const float* __restrict__ mem,
         __bf16* __restrict__ ctxcat)
{
    const int b = blockIdx.x;
    const int tid = threadIdx.x;
    float acc[4] = {0.f, 0.f, 0.f, 0.f};
    const float* mb = mem + (size_t)b * S_ * H_;
    const float* ab = attn + b * S_;
    for (int s = 0; s < S_; ++s) {
        float a = ab[s];
        const float* row = mb + (size_t)s * H_;
        #pragma unroll
        for (int j = 0; j < 4; ++j) acc[j] += a * row[tid + j * 256];
    }
    #pragma unroll
    for (int j = 0; j < 4; ++j)
        ctxcat[b * 2048 + tid + j * 256] = (__bf16)acc[j];
}

// out = tanh(tmp); bf16 feed for next step, fp32 decoder_outputs slice.
__global__ void __launch_bounds__(256)
out_tanh(const float* __restrict__ tmp, __bf16* __restrict__ feed,
         float* __restrict__ dout, int t)
{
    int idx = blockIdx.x * 256 + threadIdx.x;   // B_*H_ exact
    int b = idx >> 10, h = idx & 1023;
    float v = tanhf(tmp[idx]);
    feed[idx] = (__bf16)v;
    dout[((long)b * T_ + t) * H_ + h] = v;
}

// ---------------------------------------------------------------------------
extern "C" void kernel_launch(void* const* d_in, const int* in_sizes, int n_in,
                              void* d_out, int out_size, void* d_ws, size_t ws_size,
                              hipStream_t stream)
{
    (void)in_sizes; (void)n_in; (void)out_size; (void)ws_size;

    const int*   tgt    = (const int*)  d_in[0];
    const float* mem    = (const float*)d_in[1];
    const int*   masks  = (const int*)  d_in[2];
    const float* h0_in  = (const float*)d_in[3];   // [2,B,H]
    const float* c0_in  = (const float*)d_in[4];   // [2,B,H]
    const float* embtab = (const float*)d_in[5];
    const float* w_ih0  = (const float*)d_in[6];
    const float* w_hh0  = (const float*)d_in[7];
    const float* b_ih0  = (const float*)d_in[8];
    const float* b_hh0  = (const float*)d_in[9];
    const float* w_ih1  = (const float*)d_in[10];
    const float* w_hh1  = (const float*)d_in[11];
    const float* b_ih1  = (const float*)d_in[12];
    const float* b_hh1  = (const float*)d_in[13];
    const float* w_in   = (const float*)d_in[14];
    const float* w_out  = (const float*)d_in[15];
    float* dout = (float*)d_out;

    char* ws = (char*)d_ws;
    size_t off = 0;
    auto alloc = [&](size_t bytes) -> void* {
        off = (off + 255) & ~(size_t)255;
        void* p = ws + off;
        off += bytes;
        return p;
    };

    __bf16* Wcat0   = (__bf16*)alloc(4096ull * 2560 * 2);  // [w_ih0 | w_hh0]
    __bf16* Wcat1   = (__bf16*)alloc(4096ull * 2048 * 2);  // [w_ih1 | w_hh1]
    __bf16* Winb    = (__bf16*)alloc(1024ull * 1024 * 2);
    __bf16* Woutb   = (__bf16*)alloc(1024ull * 2048 * 2);
    __bf16* emb_all = (__bf16*)alloc((size_t)B_ * T_ * E_ * 2);
    __bf16* xcat0   = (__bf16*)alloc((size_t)B_ * 2560 * 2);
    __bf16* xcat1   = (__bf16*)alloc((size_t)B_ * 2048 * 2);
    __bf16* ctxcat  = (__bf16*)alloc((size_t)B_ * 2048 * 2);
    __bf16* h0s     = (__bf16*)alloc((size_t)B_ * H_ * 2);
    __bf16* h1b     = (__bf16*)alloc((size_t)B_ * H_ * 2);
    __bf16* feed    = (__bf16*)alloc((size_t)B_ * H_ * 2);
    float* gates0   = (float*)alloc((size_t)B_ * 4096 * 4);
    float* gates1   = (float*)alloc((size_t)B_ * 4096 * 4);
    float* c0s      = (float*)alloc((size_t)B_ * H_ * 4);
    float* c1s      = (float*)alloc((size_t)B_ * H_ * 4);
    float* q        = (float*)alloc((size_t)B_ * H_ * 4);
    float* tmpo     = (float*)alloc((size_t)B_ * H_ * 4);
    float* scores   = (float*)alloc((size_t)B_ * S_ * 4);
    float* attn     = (float*)alloc((size_t)B_ * S_ * 4);
    int*   len      = (int*)  alloc((size_t)B_ * 4);

    // One-time prep (re-run every call: deterministic, no cached state).
    cvt_cat_bf16<<<2048, 256, 0, stream>>>(w_ih0, 1536, w_hh0, 1024, Wcat0, 4096L * 2560);
    cvt_cat_bf16<<<2048, 256, 0, stream>>>(w_ih1, 1024, w_hh1, 1024, Wcat1, 4096L * 2048);
    cvt_cat_bf16<<<1024, 256, 0, stream>>>(w_in,  1024, (const float*)nullptr, 0, Winb,  1024L * 1024);
    cvt_cat_bf16<<<1024, 256, 0, stream>>>(w_out, 2048, (const float*)nullptr, 0, Woutb, 1024L * 2048);
    gather_emb<<<(B_ * T_ * E_) / 256, 256, 0, stream>>>(tgt, embtab, emb_all);
    compute_len<<<1, 32, 0, stream>>>(masks, len);

    const size_t cBytes = (size_t)B_ * H_ * sizeof(float);
    hipMemcpyAsync(c0s, c0_in,           cBytes, hipMemcpyDeviceToDevice, stream);
    hipMemcpyAsync(c1s, c0_in + B_ * H_, cBytes, hipMemcpyDeviceToDevice, stream);
    hipMemsetAsync(feed, 0, (size_t)B_ * H_ * sizeof(__bf16), stream);
    init_states<<<(B_ * H_) / 256, 256, 0, stream>>>(h0_in, h0s, xcat1);

    float* out_attn = dout + (size_t)B_ * T_ * H_;

    for (int t = 0; t < T_; ++t) {
        // Layer-0 LSTM: gates0 = [emb|feed|h0] @ [w_ih0|w_hh0]^T
        pack_xcat0<<<(B_ * 2560) / 256, 256, 0, stream>>>(emb_all, feed, h0s, xcat0, t);
        gemm_bf16_wmma<<<4096 / 128, 256, 0, stream>>>(xcat0, Wcat0, gates0, 4096, 2560);
        lstm_cell0<<<(B_ * H_) / 256, 256, 0, stream>>>(gates0, b_ih0, b_hh0, c0s, h0s, xcat1);

        // Layer-1 LSTM: gates1 = [h0n|h1] @ [w_ih1|w_hh1]^T
        gemm_bf16_wmma<<<4096 / 128, 256, 0, stream>>>(xcat1, Wcat1, gates1, 4096, 2048);
        lstm_cell1<<<(B_ * H_) / 256, 256, 0, stream>>>(gates1, b_ih1, b_hh1, c1s, h1b, xcat1, ctxcat);

        // Attention ('general'): q = h1n @ w_in^T ; scores ; softmax ; ctx
        gemm_bf16_wmma<<<1024 / 128, 256, 0, stream>>>(h1b, Winb, q, 1024, 1024);
        attn_scores<<<(B_ * S_) / 8, 256, 0, stream>>>(q, mem, len, scores);
        attn_softmax<<<B_, 256, 0, stream>>>(scores, attn, out_attn, t);
        attn_ctx<<<B_, 256, 0, stream>>>(attn, mem, ctxcat);

        // Output: tanh([ctx|h1n] @ w_out^T) -> feed + decoder_outputs[:,t,:]
        gemm_bf16_wmma<<<1024 / 128, 256, 0, stream>>>(ctxcat, Woutb, tmpo, 1024, 2048);
        out_tanh<<<(B_ * H_) / 256, 256, 0, stream>>>(tmpo, feed, dout, t);
    }
}
